// SpikingOpticalFlowBranch_27616639713868
// MI455X (gfx1250) — matrix operand order repaired and checked by reading, not compiled, not run on hardware
//
#include <hip/hip_runtime.h>

#define T_STEPS 19   // T-1 flow frames
#define NB      16   // batch
#define HH      128
#define WW      128
#define CINC    2
#define CH      64
#define NOUT    10
#define BETA    0.9f
#define THRV    1.0f

#define TILE_H  8
#define TILE_W  16
#define HALO_H  10
#define HALO_W  18
#define FP      20   // padded flow row stride (half2 units)

typedef __attribute__((ext_vector_type(16))) _Float16 v16h;
typedef __attribute__((ext_vector_type(8)))  float    v8f;

union AV { v16h v; _Float16 h[16]; unsigned u[8]; };
union H2 { unsigned u; _Float16 h[2]; };

// ---------------------------------------------------------------------------
// Kernel 0: zero the per-(t,b,ch) spike-count accumulator in workspace
// ---------------------------------------------------------------------------
__global__ void sof_zero_kernel(float* __restrict__ p, int n) {
    int i = blockIdx.x * blockDim.x + threadIdx.x;
    if (i < n) p[i] = 0.0f;
}

// ---------------------------------------------------------------------------
// Kernel 1: flow + 3x3 conv (implicit GEMM via v_wmma_f32_16x16x32_f16) + LIF
// Block = 256 threads = 8 waves; each wave owns one 16-pixel row segment and
// keeps V (16px x 64ch) resident in 4 v8f WMMA accumulators across all steps.
// Grid: (W/16, H/8, B)
// ---------------------------------------------------------------------------
__global__ __launch_bounds__(256)
void sof_conv_lif_kernel(const float* __restrict__ x,
                         const float* __restrict__ wconv,
                         const float* __restrict__ bconv,
                         float* __restrict__ g_scnt)
{
    __shared__ float    xs[2][HALO_H * HALO_W * CINC]; // ping-pong x tiles (f32)
    __shared__ unsigned flowh[HALO_H * FP];            // flow as half2 (cin0,cin1)
    __shared__ float    scnt[8][CH];                   // per-wave channel spike sums

    const int tid  = threadIdx.x;
    const int lane = tid & 31;
    const int wave = tid >> 5;          // tile row 0..7
    const int b    = blockIdx.z;
    const int row0 = blockIdx.y * TILE_H;
    const int col0 = blockIdx.x * TILE_W;
    const int m    = lane & 15;         // pixel column within tile (WMMA M)
    const int half = lane >> 4;         // lane half selects K range

    // ---- Pack conv weights into WMMA B layout (32x16 f16 per 16-ch group).
    // B element e of lane L holds K = (e<8 ? (L>>4)*8+e : 16+(L>>4)*8+e-8), N = L&15;  K = tap*2+cin,
    // tap = ky*3+kx, so memory index is simply k*CH + ch. K>=18 padded zero.
    v16h breg[4];
    #pragma unroll
    for (int nt = 0; nt < 4; ++nt) {
        AV bv;
        #pragma unroll
        for (int e = 0; e < 16; ++e) {
            int k = (e < 8) ? (half * 8 + e) : (16 + half * 8 + (e - 8));
            float wv = (k < 18) ? wconv[k * CH + nt * 16 + m] : 0.0f;
            bv.h[e] = (_Float16)wv;
        }
        breg[nt] = bv.v;
    }
    float bc[4];
    #pragma unroll
    for (int nt = 0; nt < 4; ++nt) bc[nt] = bconv[nt * 16 + m];

    // ---- Membrane potential in WMMA C layout, resident for all 19 steps
    v8f V[4];
    #pragma unroll
    for (int nt = 0; nt < 4; ++nt)
        #pragma unroll
        for (int j = 0; j < 8; ++j) V[nt][j] = 0.0f;

    const float2* x2 = (const float2*)x;   // CIN==2 -> one float2 per pixel
    int hr = 0, hc = 0, gr = 0, gc = 0, inb = 0;
    if (tid < HALO_H * HALO_W) {
        hr = tid / HALO_W; hc = tid % HALO_W;
        gr = row0 - 1 + hr; gc = col0 - 1 + hc;
        inb = ((unsigned)gr < HH) && ((unsigned)gc < WW);
    }

    // preload frame 0 into buffer 0
    if (tid < HALO_H * HALO_W) {
        float v0 = 0.0f, v1 = 0.0f;
        if (inb) { float2 p = x2[(size_t)((0 * NB + b) * HH + gr) * WW + gc]; v0 = p.x; v1 = p.y; }
        xs[0][tid * 2 + 0] = v0; xs[0][tid * 2 + 1] = v1;
    }

    for (int t = 0; t < T_STEPS; ++t) {
        // load frame t+1 into the other buffer; prefetch frame t+2
        if (tid < HALO_H * HALO_W) {
            float v0 = 0.0f, v1 = 0.0f;
            if (inb) {
                float2 p = x2[(size_t)(((t + 1) * NB + b) * HH + gr) * WW + gc];
                v0 = p.x; v1 = p.y;
                if (t + 2 <= T_STEPS)
                    __builtin_prefetch(&x2[(size_t)(((t + 2) * NB + b) * HH + gr) * WW + gc], 0, 1);
            }
            xs[(t + 1) & 1][tid * 2 + 0] = v0;
            xs[(t + 1) & 1][tid * 2 + 1] = v1;
        }
        __syncthreads();

        // flow = x[t+1]-x[t], converted to packed f16 pairs (both cin of a tap)
        if (tid < HALO_H * HALO_W) {
            const float* xp = xs[t & 1];
            const float* xn = xs[(t + 1) & 1];
            H2 p;
            p.h[0] = (_Float16)(xn[tid * 2 + 0] - xp[tid * 2 + 0]);
            p.h[1] = (_Float16)(xn[tid * 2 + 1] - xp[tid * 2 + 1]);
            flowh[hr * FP + hc] = p.u;
        }
        __syncthreads();

        // ---- Gather A (16x32 f16). Element pair i holds both cin of one tap.
        // half=0: pairs0-3 = taps(0,0)(0,1)(0,2)(1,0); pair4 = tap(2,2); rest 0
        // half=1: pairs0-3 = taps(1,1)(1,2)(2,0)(2,1); rest 0
        AV av;
        {
            const int base = wave * FP + m;     // halo row = wave+dy, col = m+dx
            unsigned o0 = half ? (unsigned)(FP + 1)     : 0u;
            unsigned o1 = half ? (unsigned)(FP + 2)     : 1u;
            unsigned o2 = half ? (unsigned)(2 * FP)     : 2u;
            unsigned o3 = half ? (unsigned)(2 * FP + 1) : (unsigned)FP;
            av.u[0] = flowh[base + o0];
            av.u[1] = flowh[base + o1];
            av.u[2] = flowh[base + o2];
            av.u[3] = flowh[base + o3];
            av.u[4] = half ? 0u : flowh[base + 2 * FP + 2];
            av.u[5] = 0u; av.u[6] = 0u; av.u[7] = 0u;
        }

        // ---- V = beta*V + b_conv + conv(flow)  via WMMA; fire & soft reset
        float csum[4];
        #pragma unroll
        for (int nt = 0; nt < 4; ++nt) {
            v8f c = V[nt] * BETA + bc[nt];
            v8f d = __builtin_amdgcn_wmma_f32_16x16x32_f16(
                        false, av.v, false, breg[nt], (short)0, c, false, false);
            float cs = 0.0f;
            #pragma unroll
            for (int j = 0; j < 8; ++j) {
                float vv = d[j];
                float s  = (vv > THRV) ? 1.0f : 0.0f;
                d[j] = vv - s * THRV;
                cs  += s;
            }
            V[nt]    = d;
            csum[nt] = cs;
        }

        // ---- per-channel spike sums: lanes L and L+16 share channel (lane&15)
        #pragma unroll
        for (int nt = 0; nt < 4; ++nt) {
            float cs = csum[nt];
            cs += __shfl_xor(cs, 16, 32);
            if (half == 0) scnt[wave][nt * 16 + m] = cs;
        }
        __syncthreads();
        if (tid < CH) {
            float tot = 0.0f;
            #pragma unroll
            for (int w8 = 0; w8 < 8; ++w8) tot += scnt[w8][tid];
            atomicAdd(&g_scnt[((size_t)t * NB + b) * CH + tid], tot);
        }
        // next iteration's first __syncthreads() protects flowh/scnt reuse
    }
}

// ---------------------------------------------------------------------------
// Kernel 2: head GEMV (64x10), time-mean readout, global spike rate
// out = [ readout 16*10 | logits_seq 19*16*10 | sr 1 ]
// ---------------------------------------------------------------------------
__global__ __launch_bounds__(512)
void sof_head_kernel(const float* __restrict__ g_scnt,
                     const float* __restrict__ whead,
                     const float* __restrict__ bhead,
                     float* __restrict__ out)
{
    __shared__ float llog[T_STEPS * NB * NOUT]; // 3040
    __shared__ float ssum[T_STEPS * NB];        // 304
    const int tid = threadIdx.x;

    if (tid < T_STEPS * NB) {
        const float inv_hw = 1.0f / (float)(HH * WW);
        float acc[NOUT];
        #pragma unroll
        for (int o = 0; o < NOUT; ++o) acc[o] = bhead[o];
        float tot = 0.0f;
        const float* sc = &g_scnt[(size_t)tid * CH];
        for (int ch = 0; ch < CH; ++ch) {
            float v  = sc[ch];
            tot     += v;
            float vm = v * inv_hw;
            #pragma unroll
            for (int o = 0; o < NOUT; ++o) acc[o] += vm * whead[ch * NOUT + o];
        }
        ssum[tid] = tot;
        #pragma unroll
        for (int o = 0; o < NOUT; ++o) {
            llog[tid * NOUT + o]            = acc[o];
            out[NB * NOUT + tid * NOUT + o] = acc[o];   // logits_seq
        }
    }
    __syncthreads();

    if (tid < NB * NOUT) {   // readout = mean over time
        int bb = tid / NOUT, o = tid % NOUT;
        float s = 0.0f;
        for (int t = 0; t < T_STEPS; ++t) s += llog[(t * NB + bb) * NOUT + o];
        out[tid] = s / (float)T_STEPS;
    }
    if (tid == 0) {          // spike rate scalar
        float s = 0.0f;
        for (int i = 0; i < T_STEPS * NB; ++i) s += ssum[i];
        out[NB * NOUT + T_STEPS * NB * NOUT] =
            s / ((float)T_STEPS * (float)NB * (float)HH * (float)WW * (float)CH);
    }
}

// ---------------------------------------------------------------------------
extern "C" void kernel_launch(void* const* d_in, const int* in_sizes, int n_in,
                              void* d_out, int out_size, void* d_ws, size_t ws_size,
                              hipStream_t stream)
{
    const float* x     = (const float*)d_in[0];   // [20,16,128,128,2]
    const float* wconv = (const float*)d_in[1];   // [3,3,2,64]
    const float* bconv = (const float*)d_in[2];   // [64]
    const float* whead = (const float*)d_in[3];   // [64,10]
    const float* bhead = (const float*)d_in[4];   // [10]
    float* g_scnt = (float*)d_ws;                 // [19,16,64] spike sums

    const int n = T_STEPS * NB * CH;
    sof_zero_kernel<<<(n + 255) / 256, 256, 0, stream>>>(g_scnt, n);

    dim3 grid(WW / TILE_W, HH / TILE_H, NB);      // (8,16,16)
    sof_conv_lif_kernel<<<grid, 256, 0, stream>>>(x, wconv, bconv, g_scnt);

    sof_head_kernel<<<1, 512, 0, stream>>>(g_scnt, whead, bhead, (float*)d_out);
}
